// GroupedMaskMultiHeadAttention_5042291605896
// MI455X (gfx1250) — compile-verified, matrix operand order checked by reading, not run on hardware
//
#include <hip/hip_runtime.h>
#include <hip/hip_bf16.h>

typedef __attribute__((ext_vector_type(16))) _Float16 v16h;
typedef __attribute__((ext_vector_type(8)))  float    v8f;
typedef __attribute__((ext_vector_type(4)))  float    f32x4;
typedef __attribute__((ext_vector_type(4)))  int      v4i;

#define D_MODEL  1536
#define HEAD_DIM 128
#define HG       4
#define NB       2
#define SEQ      2048
#define MROWS    (NB*SEQ)      /* 4096 */
#define PDIM     (HG*HEAD_DIM) /* 512  */
#define EPSF     1.1920928955078125e-07f

#if defined(__AMDGCN__) && __has_builtin(__builtin_amdgcn_global_load_async_to_lds_b128)
#define HAVE_ASYNC_LDS 1
#else
#define HAVE_ASYNC_LDS 0
#endif

typedef __attribute__((address_space(1))) v4i gv4i;  // global int4
typedef __attribute__((address_space(3))) v4i lv4i;  // LDS int4

__device__ __forceinline__ void async_cp16(const void* g, void* l) {
#if HAVE_ASYNC_LDS
  __builtin_amdgcn_global_load_async_to_lds_b128((gv4i*)g, (lv4i*)l, 0, 0);
#endif
}

__device__ __forceinline__ void wait_async0() {
#if __has_builtin(__builtin_amdgcn_s_wait_asynccnt)
  __builtin_amdgcn_s_wait_asynccnt(0);
#else
  asm volatile("s_wait_asynccnt 0" ::: "memory");
#endif
}

union Pack8 { _Float16 h[8]; uint4 u; };

__device__ __forceinline__ v8f wmma_f16(v16h a, v16h b, v8f c) {
  return __builtin_amdgcn_wmma_f32_16x16x32_f16(false, a, false, b, (short)0, c,
                                                false, false);
}

// ---------------------------------------------------------------------------
// C[M,N] = A[M,K] @ W[N,K]^T   (A: f32 or f16, W: f32, C: f32)
// Block: 128x64 tile, 256 threads (8 waves), K-step 32.
// LDS tiles stored pre-swizzled into the CDNA5 16-bit WMMA fragment layout:
//   lane = (row&15) + 16*((k>>3)&1),  half = (k&7) + 8*(k>>4)
// so each lane's v16h fragment is one contiguous 32B LDS read.
// f16 A-tiles are moved with GLOBAL_LOAD_ASYNC_TO_LDS_B128 when available.
// ---------------------------------------------------------------------------
template <typename AT>
__global__ __launch_bounds__(256)
void gemm_xwT(const AT* __restrict__ A, const float* __restrict__ W,
              float* __restrict__ C, int M, int N, int K) {
  __shared__ alignas(16) _Float16 As[8][32][16]; // 8 M-subtiles
  __shared__ alignas(16) _Float16 Bs[4][32][16]; // 4 N-subtiles

  const int tid  = threadIdx.x;
  const int lane = tid & 31;
  const int wave = tid >> 5;
  const int wm   = wave >> 1;   // 0..3 : M
  const int wn   = wave & 1;    // 0..1 : N
  const int m0   = blockIdx.y * 128;
  const int n0   = blockIdx.x * 64;

  v8f acc[2][2] = {};

  for (int k0 = 0; k0 < K; k0 += 32) {
    // ---- stage A tile (128x32) ----
#pragma unroll
    for (int i = 0; i < 2; ++i) {
      int g   = tid + i * 256;         // 0..511
      int m   = g >> 2;
      int kg  = (g & 3) << 3;          // 0,8,16,24
      int sub = m >> 4;
      int ln  = (m & 15) + ((kg & 8) ? 16 : 0);
      int h0  = (kg & 16) ? 8 : 0;
      if constexpr (sizeof(AT) == 2) {
#if HAVE_ASYNC_LDS
        async_cp16(A + (size_t)(m0 + m) * K + k0 + kg, &As[sub][ln][h0]);
#else
        *(uint4*)&As[sub][ln][h0] =
            *(const uint4*)(A + (size_t)(m0 + m) * K + k0 + kg);
#endif
      } else {
        const f32x4* src = (const f32x4*)(A + (size_t)(m0 + m) * K + k0 + kg);
        f32x4 f0 = src[0], f1 = src[1];
        Pack8 p;
        p.h[0] = (_Float16)f0.x; p.h[1] = (_Float16)f0.y;
        p.h[2] = (_Float16)f0.z; p.h[3] = (_Float16)f0.w;
        p.h[4] = (_Float16)f1.x; p.h[5] = (_Float16)f1.y;
        p.h[6] = (_Float16)f1.z; p.h[7] = (_Float16)f1.w;
        *(uint4*)&As[sub][ln][h0] = p.u;
      }
    }
    // ---- stage B tile (64x32 rows of W) ----
    {
      int g   = tid;                   // 0..255
      int n   = g >> 2;
      int kg  = (g & 3) << 3;
      int sub = n >> 4;
      int ln  = (n & 15) + ((kg & 8) ? 16 : 0);
      int h0  = (kg & 16) ? 8 : 0;
      const f32x4* src = (const f32x4*)(W + (size_t)(n0 + n) * K + k0 + kg);
      f32x4 f0 = src[0], f1 = src[1];
      Pack8 p;
      p.h[0] = (_Float16)f0.x; p.h[1] = (_Float16)f0.y;
      p.h[2] = (_Float16)f0.z; p.h[3] = (_Float16)f0.w;
      p.h[4] = (_Float16)f1.x; p.h[5] = (_Float16)f1.y;
      p.h[6] = (_Float16)f1.z; p.h[7] = (_Float16)f1.w;
      *(uint4*)&Bs[sub][ln][h0] = p.u;
    }
#if HAVE_ASYNC_LDS
    if constexpr (sizeof(AT) == 2) wait_async0();
#endif
    __syncthreads();

    v16h a0 = *(const v16h*)As[wm * 2 + 0][lane];
    v16h a1 = *(const v16h*)As[wm * 2 + 1][lane];
    v16h b0 = *(const v16h*)Bs[wn * 2 + 0][lane];
    v16h b1 = *(const v16h*)Bs[wn * 2 + 1][lane];
    acc[0][0] = wmma_f16(a0, b0, acc[0][0]);
    acc[0][1] = wmma_f16(a0, b1, acc[0][1]);
    acc[1][0] = wmma_f16(a1, b0, acc[1][0]);
    acc[1][1] = wmma_f16(a1, b1, acc[1][1]);
    __syncthreads();
  }

  // epilogue: C layout -> lane holds col (lane&15), rows 8*(lane>>4)+r
  const int hi = lane >> 4;
  const int nl = lane & 15;
#pragma unroll
  for (int sm = 0; sm < 2; ++sm)
#pragma unroll
    for (int sn = 0; sn < 2; ++sn)
#pragma unroll
      for (int r = 0; r < 8; ++r) {
        int mg = m0 + wm * 32 + sm * 16 + hi * 8 + r;
        int ng = n0 + wn * 32 + sn * 16 + nl;
        C[(size_t)mg * N + ng] = acc[sm][sn][r];
      }
}

// ---------------------------------------------------------------------------
// RMSNorm + RoPE (q,k) or plain convert (v).
// P: [B*T, HG*128] f32 -> dst: [B, HG, T, 128] f16. One wave per (row, head).
// ---------------------------------------------------------------------------
__global__ __launch_bounds__(256)
void qkv_transform(const float* __restrict__ P, _Float16* __restrict__ dst,
                   const float* __restrict__ nw, int do_rope) {
  int gw   = (int)((blockIdx.x * 256 + threadIdx.x) >> 5);
  int lane = threadIdx.x & 31;
  int h    = gw & 3;
  int row  = gw >> 2;
  if (row >= MROWS) return;
  int b = row / SEQ;
  int t = row - b * SEQ;

  const float* src = P + (size_t)row * PDIM + h * HEAD_DIM;
  float x0 = src[lane], x1 = src[lane + 32], x2 = src[lane + 64], x3 = src[lane + 96];
  float y0, y1, y2, y3;

  if (do_rope) {
    float ss = x0 * x0 + x1 * x1 + x2 * x2 + x3 * x3;
#pragma unroll
    for (int m = 16; m >= 1; m >>= 1) ss += __shfl_xor(ss, m, 32);
    float r = rsqrtf(ss * (1.0f / 128.0f) + EPSF);
    x0 *= r * nw[lane];      x1 *= r * nw[lane + 32];
    x2 *= r * nw[lane + 64]; x3 *= r * nw[lane + 96];
    const float LOGT = 13.815510557964274f;  // ln(1e6)
    float inv0 = __expf(-((float)(2 * lane)        * (1.0f / 128.0f)) * LOGT);
    float inv1 = __expf(-((float)(2 * (lane + 32)) * (1.0f / 128.0f)) * LOGT);
    float a0 = (float)t * inv0, a1 = (float)t * inv1;
    float c0, s0, c1, s1;
    __sincosf(a0, &s0, &c0);
    __sincosf(a1, &s1, &c1);
    y0 = x0 * c0 - x2 * s0;
    y2 = x2 * c0 + x0 * s0;
    y1 = x1 * c1 - x3 * s1;
    y3 = x3 * c1 + x1 * s1;
  } else {
    y0 = x0; y1 = x1; y2 = x2; y3 = x3;
  }
  _Float16* d = dst + (((size_t)b * HG + h) * SEQ + t) * HEAD_DIM;
  d[lane]      = (_Float16)y0;
  d[lane + 32] = (_Float16)y1;
  d[lane + 64] = (_Float16)y2;
  d[lane + 96] = (_Float16)y3;
}

// ---------------------------------------------------------------------------
// Flash attention, one block per (branch fixed, b, h, 64-row q tile).
// 4 waves x 16 q-rows. QK^T and P*V via WMMA; online softmax in C layout.
// Q/K tiles staged with async global->LDS copies; next K/V block prefetched.
// ---------------------------------------------------------------------------
__global__ __launch_bounds__(128)
void attn_fa(const _Float16* __restrict__ qh, const _Float16* __restrict__ kh,
             const _Float16* __restrict__ vh, _Float16* __restrict__ ac,
             int branch) {
  __shared__ alignas(16) _Float16 Qs[4][4][32][16]; // [msub][kstep] A-frags
  __shared__ alignas(16) _Float16 Ks[4][2][32][16]; // [kstep][jsub] B-frags
  __shared__ alignas(16) _Float16 Vs[8][32][16];    // [dsub]        B-frags
  __shared__ alignas(16) _Float16 Ps[4][32][16];    // per-wave P    A-frag

  const int tid  = threadIdx.x;
  const int lane = tid & 31;
  const int w    = tid >> 5;
  const int qt   = blockIdx.x & 31;
  const int h    = (blockIdx.x >> 5) & 3;
  const int b    = blockIdx.x >> 7;
  const int q0   = qt * 64;
  const size_t base = (((size_t)b * HG) + h) * (size_t)SEQ * HEAD_DIM;
  const _Float16* Q  = qh + base;
  const _Float16* Kp = kh + base;
  const _Float16* Vp = vh + base;

  // stage Q tile 64x128 into A-fragment layout
#pragma unroll
  for (int i = 0; i < 8; ++i) {
    int g  = tid + i * 128;        // 0..1023
    int m  = g >> 4;
    int dg = (g & 15) << 3;
    int ms = m >> 4;
    int ks = dg >> 5;
    int dk = dg & 31;
    int ln = (m & 15) + ((dk & 8) ? 16 : 0);
    int h0 = (dk & 16) ? 8 : 0;
#if HAVE_ASYNC_LDS
    async_cp16(Q + (size_t)(q0 + m) * HEAD_DIM + dg, &Qs[ms][ks][ln][h0]);
#else
    *(uint4*)&Qs[ms][ks][ln][h0] =
        *(const uint4*)(Q + (size_t)(q0 + m) * HEAD_DIM + dg);
#endif
  }
#if HAVE_ASYNC_LDS
  wait_async0();
#endif
  __syncthreads();
  v16h qf[4];
#pragma unroll
  for (int ks = 0; ks < 4; ++ks) qf[ks] = *(const v16h*)Qs[w][ks][lane];

  v8f o[8] = {};
  float mrow[8], lrow[8];
#pragma unroll
  for (int r = 0; r < 8; ++r) { mrow[r] = -1e30f; lrow[r] = 0.0f; }

  int kb0 = 0, kb1 = SEQ / 32;
  if (branch == 0) kb1 = q0 / 32 + 2;        // causal: j <= qi
  else if (branch == 1) kb0 = q0 / 32;       // anti-causal: j >= qi

  const int hi = lane >> 4;
  const int nl = lane & 15;
  const float scale = 0.08838834764831845f;  // 1/sqrt(128)

  for (int kb = kb0; kb < kb1; ++kb) {
    const int jb = kb * 32;
    __syncthreads();  // previous iteration's LDS readers are done

    // prefetch next K/V block (128 threads x 64B cover the 8KB blocks)
    if (kb + 1 < kb1) {
      __builtin_prefetch(Kp + (size_t)(jb + 32) * HEAD_DIM + tid * 32, 0, 3);
      __builtin_prefetch(Vp + (size_t)(jb + 32) * HEAD_DIM + tid * 32, 0, 3);
    }

    // stage K block 32x128 -> B-fragments for QK^T (async copies)
#pragma unroll
    for (int i = 0; i < 4; ++i) {
      int g  = tid + i * 128;      // 0..511
      int j  = g >> 4;
      int dg = (g & 15) << 3;
      int ks = dg >> 5;
      int js = j >> 4;
      int dk = dg & 31;
      int ln = (j & 15) + ((dk & 8) ? 16 : 0);
      int h0 = (dk & 16) ? 8 : 0;
#if HAVE_ASYNC_LDS
      async_cp16(Kp + (size_t)(jb + j) * HEAD_DIM + dg, &Ks[ks][js][ln][h0]);
#else
      *(uint4*)&Ks[ks][js][ln][h0] =
          *(const uint4*)(Kp + (size_t)(jb + j) * HEAD_DIM + dg);
#endif
    }
    // stage V block 32x128 -> B-fragments for P*V (element transpose in VALU)
#pragma unroll
    for (int i = 0; i < 4; ++i) {
      int g  = tid + i * 128;
      int j  = g >> 4;
      int dg = (g & 15) << 3;
      Pack8 p;
      p.u = *(const uint4*)(Vp + (size_t)(jb + j) * HEAD_DIM + dg);
      int hh = (j & 7) + ((j & 16) ? 8 : 0);
      int la = (j & 8) ? 16 : 0;
#pragma unroll
      for (int e = 0; e < 8; ++e) {
        int d = dg + e;
        Vs[d >> 4][(d & 15) + la][hh] = p.h[e];
      }
    }
#if HAVE_ASYNC_LDS
    wait_async0();
#endif
    __syncthreads();

    // S = Q @ K^T
    v8f s[2] = {};
#pragma unroll
    for (int js = 0; js < 2; ++js)
#pragma unroll
      for (int ks = 0; ks < 4; ++ks)
        s[js] = wmma_f16(qf[ks], *(const v16h*)Ks[ks][js][lane], s[js]);

    // scale + analytic mask
#pragma unroll
    for (int js = 0; js < 2; ++js)
#pragma unroll
      for (int r = 0; r < 8; ++r) {
        int mg = q0 + w * 16 + hi * 8 + r;
        int jg = jb + js * 16 + nl;
        float sv = s[js][r] * scale;
        if ((branch == 0 && jg > mg) || (branch == 1 && jg < mg)) sv = -1e30f;
        s[js][r] = sv;
      }

    // online softmax (rows live across the 16 lanes of each half)
#pragma unroll
    for (int r = 0; r < 8; ++r) {
      float rm = fmaxf(s[0][r], s[1][r]);
#pragma unroll
      for (int m = 8; m >= 1; m >>= 1) rm = fmaxf(rm, __shfl_xor(rm, m, 32));
      float mn   = fmaxf(mrow[r], rm);
      float corr = __expf(mrow[r] - mn);
      float p0   = __expf(s[0][r] - mn);
      float p1   = __expf(s[1][r] - mn);
      float rs   = p0 + p1;
#pragma unroll
      for (int m = 8; m >= 1; m >>= 1) rs += __shfl_xor(rs, m, 32);
      lrow[r] = lrow[r] * corr + rs;
      mrow[r] = mn;
      s[0][r] = p0;
      s[1][r] = p1;
#pragma unroll
      for (int ds = 0; ds < 8; ++ds) o[ds][r] *= corr;
    }

    // P (C layout) -> per-wave LDS patch in A-fragment layout
#pragma unroll
    for (int js = 0; js < 2; ++js)
#pragma unroll
      for (int r = 0; r < 8; ++r) {
        int ml = hi * 8 + r;       // 0..15
        int jl = js * 16 + nl;     // 0..31
        int ln = ml + ((jl & 8) ? 16 : 0);
        int hh = (jl & 7) + ((jl & 16) ? 8 : 0);
        Ps[w][ln][hh] = (_Float16)s[js][r];
      }
    asm volatile("s_wait_dscnt 0" ::: "memory");  // wave-local LDS visibility
    v16h pf = *(const v16h*)Ps[w][lane];
#pragma unroll
    for (int ds = 0; ds < 8; ++ds)
      o[ds] = wmma_f16(pf, *(const v16h*)Vs[ds][lane], o[ds]);
  }

  // normalize and write into concat buffer [B, T, 12, 128] (f16)
#pragma unroll
  for (int ds = 0; ds < 8; ++ds)
#pragma unroll
    for (int r = 0; r < 8; ++r) {
      int tg = q0 + w * 16 + hi * 8 + r;
      float ov = o[ds][r] / lrow[r];
      size_t idx = (((size_t)b * SEQ + tg) * 12 + branch * 4 + h) * HEAD_DIM +
                   ds * 16 + nl;
      ac[idx] = (_Float16)ov;
    }
}

// ---------------------------------------------------------------------------
extern "C" void kernel_launch(void* const* d_in, const int* in_sizes, int n_in,
                              void* d_out, int out_size, void* d_ws,
                              size_t ws_size, hipStream_t stream) {
  const float* X   = (const float*)d_in[0];
  const float* wq[3] = {(const float*)d_in[1], (const float*)d_in[4], (const float*)d_in[7]};
  const float* wk[3] = {(const float*)d_in[2], (const float*)d_in[5], (const float*)d_in[8]};
  const float* wv[3] = {(const float*)d_in[3], (const float*)d_in[6], (const float*)d_in[9]};
  const float* wo  = (const float*)d_in[10];
  const float* qnw = (const float*)d_in[11];
  const float* knw = (const float*)d_in[12];
  // d_in[13..15]: masks, handled analytically per branch

  char* ws = (char*)d_ws;
  float*     P   = (float*)ws;      ws += (size_t)MROWS * PDIM * 4;     // 8 MiB
  _Float16*  qhb = (_Float16*)ws;   ws += (size_t)MROWS * PDIM * 2;     // 4 MiB
  _Float16*  khb = (_Float16*)ws;   ws += (size_t)MROWS * PDIM * 2;     // 4 MiB
  _Float16*  vhb = (_Float16*)ws;   ws += (size_t)MROWS * PDIM * 2;     // 4 MiB
  _Float16*  ac  = (_Float16*)ws;   ws += (size_t)MROWS * D_MODEL * 2;  // 12 MiB

  const dim3 gemmBlk(256);
  const dim3 gemmGridP(PDIM / 64, MROWS / 128);     // (8, 32)
  const dim3 gemmGridO(D_MODEL / 64, MROWS / 128);  // (24, 32)
  const int  xfBlocks = (MROWS * HG) / 8;           // 2048 blocks of 8 waves
  const int  attnBlocks = NB * HG * (SEQ / 64);     // 256

  for (int br = 0; br < 3; ++br) {
    gemm_xwT<float><<<gemmGridP, gemmBlk, 0, stream>>>(X, wq[br], P, MROWS, PDIM, D_MODEL);
    qkv_transform<<<xfBlocks, 256, 0, stream>>>(P, qhb, qnw, 1);

    gemm_xwT<float><<<gemmGridP, gemmBlk, 0, stream>>>(X, wk[br], P, MROWS, PDIM, D_MODEL);
    qkv_transform<<<xfBlocks, 256, 0, stream>>>(P, khb, knw, 1);

    gemm_xwT<float><<<gemmGridP, gemmBlk, 0, stream>>>(X, wv[br], P, MROWS, PDIM, D_MODEL);
    qkv_transform<<<xfBlocks, 256, 0, stream>>>(P, vhb, qnw, 0);

    attn_fa<<<attnBlocks, 128, 0, stream>>>(qhb, khb, vhb, ac, br);
  }
  gemm_xwT<_Float16><<<gemmGridO, gemmBlk, 0, stream>>>(ac, wo, (float*)d_out,
                                                        MROWS, D_MODEL, D_MODEL);
}
